// ConvAttention_76364518523159
// MI455X (gfx1250) — matrix operand order repaired and checked by reading, not compiled, not run on hardware
//
#include <hip/hip_runtime.h>
#include <hip/hip_bf16.h>

typedef __attribute__((ext_vector_type(16))) _Float16 v16h;
typedef __attribute__((ext_vector_type(8)))  float    v8f;

#define SCALE_C (-0.0005f)
#define EPS_P   (1e-8f)

__device__ __forceinline__ void cvt4(_Float16* h, float4 v) {
    h[0] = (_Float16)v.x; h[1] = (_Float16)v.y;
    h[2] = (_Float16)v.z; h[3] = (_Float16)v.w;
}
__device__ __forceinline__ float ld_f(const float* p)    { return *p; }
__device__ __forceinline__ float ld_f(const _Float16* p) { return (float)*p; }

// Stage A tile (128 x 32) into LDS [m][k], k-contiguous. Thread t: row t>>1, 16 halves.
template<typename AT, bool AKS1, bool ASYNCA>
__device__ __forceinline__ void stage_A(
    _Float16* __restrict__ dstA, const AT* __restrict__ Ab,
    int a_ms, int a_ks, int m0, int kkk, int d, int M, int K, int tid)
{
    const int ml  = tid >> 1;
    const int kl0 = (tid & 1) * 16;
    const int gm  = m0 + ml;
    if (AKS1 && (kkk + 32 <= K)) {
        if (gm < M) {
            const AT* p = Ab + gm * a_ms + kkk + kl0;
            if (ASYNCA && sizeof(AT) == 2) {
                // contiguous global -> LDS copy: CDNA5 async DMA, tracked by ASYNCcnt
                unsigned lo = (unsigned)(size_t)(dstA + ml * 32 + kl0);
                asm volatile("global_load_async_to_lds_b128 %0, %1, off"
                             :: "v"(lo), "v"(p) : "memory");
                asm volatile("global_load_async_to_lds_b128 %0, %1, off"
                             :: "v"(lo + 16), "v"(p + 8) : "memory");
            } else {
                _Float16 h[16];
                if (sizeof(AT) == 2) {
                    ((uint4*)h)[0] = ((const uint4*)p)[0];
                    ((uint4*)h)[1] = ((const uint4*)p)[1];
                } else {
                    cvt4(h + 0,  ((const float4*)p)[0]);
                    cvt4(h + 4,  ((const float4*)p)[1]);
                    cvt4(h + 8,  ((const float4*)p)[2]);
                    cvt4(h + 12, ((const float4*)p)[3]);
                }
                ((uint4*)(dstA + ml * 32 + kl0))[0] = ((uint4*)h)[0];
                ((uint4*)(dstA + ml * 32 + kl0))[1] = ((uint4*)h)[1];
            }
        } else {
            uint4 z = {0u, 0u, 0u, 0u};
            ((uint4*)(dstA + ml * 32 + kl0))[0] = z;
            ((uint4*)(dstA + ml * 32 + kl0))[1] = z;
        }
    } else {
        const int cm = gm < M ? gm : (M - 1);
        #pragma unroll
        for (int i = 0; i < 16; ++i) {
            int gk = kkk + kl0 + i;
            int ck = gk < K ? gk : (K - 1);
            bool ok = (gm < M) && (gk < K);
            float v = ld_f(Ab + cm * a_ms + ck * a_ks + d);
            dstA[ml * 32 + kl0 + i] = ok ? (_Float16)v : (_Float16)0.f;
        }
    }
}

// Stage B tile (64 x 32) into LDS [n][k], k-contiguous. Thread t: k row t>>3, 8 n values.
template<typename BT>
__device__ __forceinline__ void stage_B(
    _Float16* __restrict__ dstB, const BT* __restrict__ Bb,
    int b_ks, int n0, int kkk, int shift, int N, int K, int tid)
{
    const int kl  = tid >> 3;
    const int nl0 = (tid & 7) * 8;
    const int gk  = kkk + kl;
    _Float16 h[8];
    if ((shift == 0) && (n0 + 64 <= N) && (kkk + 32 <= K)) {
        const BT* p = Bb + gk * b_ks + n0 + nl0;
        if (sizeof(BT) == 2) {
            ((uint4*)h)[0] = *(const uint4*)p;
        } else {
            cvt4(h + 0, ((const float4*)p)[0]);
            cvt4(h + 4, ((const float4*)p)[1]);
        }
    } else {
        const int ck = gk < K ? gk : (K - 1);
        #pragma unroll
        for (int j = 0; j < 8; ++j) {
            int gn = n0 + nl0 + j + shift;
            bool ok = (gk < K) && (gn >= 0) && (gn < N);
            int cn = gn < 0 ? 0 : (gn < N ? gn : (N - 1));
            float v = ld_f(Bb + ck * b_ks + cn);
            h[j] = ok ? (_Float16)v : (_Float16)0.f;
        }
    }
    #pragma unroll
    for (int j = 0; j < 8; ++j) dstB[(nl0 + j) * 32 + kl] = h[j];
}

// ---------------------------------------------------------------------------
// LDS double-buffered WMMA GEMM (specialized per call site).
//   C[b][m][n] = epilogue( sum_d sum_k A(m,k,d) * B(b,k,n+d-NTAPS/2) )
// MODE 0 = +bias, 1 = +bias+ReLU, 2 = SCALE*(q2[m]+k2[n]-2*acc)
// Block tile 128x64, 8 waves, each 32x32 via 4x v_wmma_f32_16x16x32_f16.
// ---------------------------------------------------------------------------
template<typename AT, typename BT, typename CT, int NTAPS, int MODE, bool AKS1, bool ASYNCA>
__global__ __launch_bounds__(256)
void gemm_wmma(const AT* __restrict__ A, int a_batch, int a_ms, int a_ks,
               const BT* __restrict__ Bv, int b_batch, int b_ks,
               const float* __restrict__ biasOrQ2, const float* __restrict__ k2v,
               CT* __restrict__ Cv, int c_batch, int ldc,
               int M, int N, int K)
{
    __shared__ __align__(16) _Float16 AslBuf[2][128 * 32];
    __shared__ __align__(16) _Float16 BslBuf[2][64 * 32];

    const int b    = blockIdx.z;
    const int n0   = blockIdx.x * 64;
    const int m0   = blockIdx.y * 128;
    const int tid  = threadIdx.x;
    const int lane = tid & 31;
    const int wave = tid >> 5;
    const int lmod = lane & 15;
    const int lhi  = lane >> 4;
    const int wm0  = (wave & 3) * 32;
    const int wn0  = (wave >> 2) * 32;

    const AT* Ab = A + b * a_batch;
    const BT* Bb = Bv + b * b_batch;
    const int half = NTAPS >> 1;

    v8f acc[2][2];
    #pragma unroll
    for (int i = 0; i < 2; ++i)
        #pragma unroll
        for (int j = 0; j < 2; ++j)
            acc[i][j] = (v8f){0.f,0.f,0.f,0.f,0.f,0.f,0.f,0.f};

    const int nkc = (K + 31) >> 5;
    const int nchunks = NTAPS * nkc;

    int d_nx = 0, kk_nx = 0;
    stage_A<AT, AKS1, ASYNCA>(AslBuf[0], Ab, a_ms, a_ks, m0, kk_nx, d_nx, M, K, tid);
    stage_B<BT>(BslBuf[0], Bb, b_ks, n0, kk_nx, d_nx - half, N, K, tid);
    kk_nx += 32; if (kk_nx >= K) { kk_nx = 0; ++d_nx; }
    if (ASYNCA) asm volatile("s_wait_asynccnt 0x0" ::: "memory");
    __syncthreads();

    for (int c = 0; c < nchunks; ++c) {
        const int cur = c & 1;
        if (c + 1 < nchunks) {
            stage_A<AT, AKS1, ASYNCA>(AslBuf[1 ^ cur], Ab, a_ms, a_ks, m0, kk_nx, d_nx, M, K, tid);
            stage_B<BT>(BslBuf[1 ^ cur], Bb, b_ks, n0, kk_nx, d_nx - half, N, K, tid);
            kk_nx += 32; if (kk_nx >= K) { kk_nx = 0; ++d_nx; }
        }
        const _Float16* Asl = AslBuf[cur];
        const _Float16* Bsl = BslBuf[cur];

        v16h afrag[2], bfrag[2];
        #pragma unroll
        for (int mi = 0; mi < 2; ++mi) {
            int mrow = wm0 + mi * 16 + lmod;
            v16h a = {};
            #pragma unroll
            for (int v = 0; v < 8; ++v) {
                int k = 16 * (v >> 2) + 8 * lhi + 2 * (v & 3);
                a[2 * v]     = Asl[mrow * 32 + k];
                a[2 * v + 1] = Asl[mrow * 32 + k + 1];
            }
            afrag[mi] = a;
        }
        #pragma unroll
        for (int ni = 0; ni < 2; ++ni) {
            int ncol = wn0 + ni * 16 + lmod;
            v16h bb = {};
            #pragma unroll
            for (int v = 0; v < 8; ++v) {
                int k = 16 * (v >> 2) + 8 * lhi + 2 * (v & 3);
                bb[2 * v]     = Bsl[ncol * 32 + k];
                bb[2 * v + 1] = Bsl[ncol * 32 + k + 1];
            }
            bfrag[ni] = bb;
        }
        #pragma unroll
        for (int mi = 0; mi < 2; ++mi)
            #pragma unroll
            for (int ni = 0; ni < 2; ++ni)
                acc[mi][ni] = __builtin_amdgcn_wmma_f32_16x16x32_f16(
                    false, afrag[mi], false, bfrag[ni],
                    (short)0, acc[mi][ni], false, false);
        if (ASYNCA) asm volatile("s_wait_asynccnt 0x0" ::: "memory");
        __syncthreads();
    }

    // epilogue
    CT* Cb = Cv + b * c_batch;
    #pragma unroll
    for (int mi = 0; mi < 2; ++mi) {
        #pragma unroll
        for (int ni = 0; ni < 2; ++ni) {
            #pragma unroll
            for (int r = 0; r < 8; ++r) {
                int gm = m0 + wm0 + mi * 16 + r + 8 * lhi;
                int gn = n0 + wn0 + ni * 16 + lmod;
                if (gm < M && gn < N) {
                    float v = acc[mi][ni][r];
                    if (MODE == 2) {
                        v = SCALE_C * (biasOrQ2[b * M + gm] + k2v[b * N + gn] - 2.f * v);
                    } else {
                        v += biasOrQ2[gm];
                        if (MODE == 1) v = fmaxf(v, 0.f);
                    }
                    Cb[gm * ldc + gn] = (CT)v;
                }
            }
        }
    }
}

// keys2h[b][c][t] = f16(keys[b][c][t] + style[b][t][c])
__global__ void addstyle_kernel(const float* __restrict__ keys,
                                const float* __restrict__ style,
                                _Float16* __restrict__ out, int Cc, int T)
{
    int idx = blockIdx.x * 256 + threadIdx.x;
    int b = blockIdx.y;
    if (idx >= Cc * T) return;
    int c = idx / T, t = idx - c * T;
    float v = keys[b * Cc * T + idx] + style[(b * T + t) * Cc + c];
    out[b * Cc * T + idx] = (_Float16)v;
}

// f16 transpose: in [b][R][Cc] -> out [b][Cc][R]
__global__ __launch_bounds__(256)
void transpose_h_kernel(const _Float16* __restrict__ in, _Float16* __restrict__ out,
                        int R, int Cc, int ib, int ob)
{
    __shared__ _Float16 sh[32][33];
    int b = blockIdx.z;
    int c0 = blockIdx.x * 32, r0 = blockIdx.y * 32;
    const _Float16* ip = in + b * ib;
    _Float16* op = out + b * ob;
    for (int i = threadIdx.y; i < 32; i += 8) {
        int r = r0 + i, c = c0 + threadIdx.x;
        sh[i][threadIdx.x] = (r < R && c < Cc) ? ip[r * Cc + c] : (_Float16)0.f;
    }
    __syncthreads();
    for (int i = threadIdx.y; i < 32; i += 8) {
        int c = c0 + i, r = r0 + threadIdx.x;
        if (c < Cc && r < R) op[c * R + r] = sh[threadIdx.x][i];
    }
}

// out[b][t] = sum_c xT[b][t][c]^2   (xT is [T][Cc] f16)
__global__ void sumsq_t_kernel(const _Float16* __restrict__ xT, float* __restrict__ out,
                               int Cc, int T, int xbatch)
{
    int t = blockIdx.x * 256 + threadIdx.x;
    int b = blockIdx.y;
    if (t >= T) return;
    const _Float16* p = xT + b * xbatch + t * Cc;
    float s = 0.f;
    for (int c = 0; c < Cc; ++c) { float v = (float)p[c]; s += v * v; }
    out[b * T + t] = s;
}

__device__ __forceinline__ float block_reduce(float v, float* sh, bool ismax)
{
    #pragma unroll
    for (int o = 16; o > 0; o >>= 1) {
        float ov = __shfl_xor(v, o, 32);
        v = ismax ? fmaxf(v, ov) : (v + ov);
    }
    if ((threadIdx.x & 31) == 0) sh[threadIdx.x >> 5] = v;
    __syncthreads();
    float r = sh[0];
    #pragma unroll
    for (int i = 1; i < 4; ++i) r = ismax ? fmaxf(r, sh[i]) : (r + sh[i]);
    __syncthreads();
    return r;
}

// One block (128 thr) per (b,t1) row: scores -> logprob (in place) + masked softmax.
__global__ __launch_bounds__(128)
void row_softmax_kernel(float* __restrict__ attn, float* __restrict__ lp_io,
                        const float* __restrict__ prior,
                        const unsigned char* __restrict__ mask,
                        int T1, int T2)
{
    __shared__ float sh[4];
    const int b  = blockIdx.y;
    const int t1 = blockIdx.x;
    const size_t base = ((size_t)b * T1 + t1) * (size_t)T2;

    float sv[4];
    float lmax = -INFINITY;
    #pragma unroll
    for (int i = 0; i < 4; ++i) {
        int t2 = threadIdx.x + i * 128;
        float xv = (t2 < T2) ? lp_io[base + t2] : -INFINITY;
        sv[i] = xv;
        lmax = fmaxf(lmax, xv);
    }
    float m = block_reduce(lmax, sh, true);
    float ls = 0.f;
    #pragma unroll
    for (int i = 0; i < 4; ++i) {
        int t2 = threadIdx.x + i * 128;
        if (t2 < T2) ls += expf(sv[i] - m);
    }
    float lse = m + logf(block_reduce(ls, sh, false));

    float mv[4];
    float lmax2 = -INFINITY;
    #pragma unroll
    for (int i = 0; i < 4; ++i) {
        int t2 = threadIdx.x + i * 128;
        if (t2 < T2) {
            float lp = sv[i] - lse + logf(prior[base + t2] + EPS_P);
            lp_io[base + t2] = lp;
            mv[i] = mask[b * T2 + t2] ? -INFINITY : lp;
        } else {
            mv[i] = -INFINITY;
        }
        lmax2 = fmaxf(lmax2, mv[i]);
    }
    float m2 = block_reduce(lmax2, sh, true);
    float s2 = 0.f;
    #pragma unroll
    for (int i = 0; i < 4; ++i) {
        int t2 = threadIdx.x + i * 128;
        if (t2 < T2) s2 += expf(mv[i] - m2);
    }
    float tot = block_reduce(s2, sh, false);
    #pragma unroll
    for (int i = 0; i < 4; ++i) {
        int t2 = threadIdx.x + i * 128;
        if (t2 < T2) attn[base + t2] = expf(mv[i] - m2) / tot;
    }
}

extern "C" void kernel_launch(void* const* d_in, const int* in_sizes, int n_in,
                              void* d_out, int out_size, void* d_ws, size_t ws_size,
                              hipStream_t stream)
{
    (void)in_sizes; (void)n_in; (void)out_size; (void)ws_size;
    const int Bc = 32, T1 = 2000, T2 = 400;
    const int NMEL = 80, NTXT = 512, NATT = 80;

    const float* queries = (const float*)d_in[0];
    const float* keys    = (const float*)d_in[1];
    const unsigned char* mask = (const unsigned char*)d_in[3];
    const float* prior   = (const float*)d_in[4];
    const float* style   = (const float*)d_in[5];
    const float* qw1 = (const float*)d_in[6];   const float* qb1 = (const float*)d_in[7];
    const float* qw2 = (const float*)d_in[8];   const float* qb2 = (const float*)d_in[9];
    const float* qw3 = (const float*)d_in[10];  const float* qb3 = (const float*)d_in[11];
    const float* kw1 = (const float*)d_in[12];  const float* kb1 = (const float*)d_in[13];
    const float* kw2 = (const float*)d_in[14];  const float* kb2 = (const float*)d_in[15];

    char* w = (char*)d_ws;
    auto alloc = [&](size_t bytes) { char* p = w; w += (bytes + 255) & ~(size_t)255; return p; };
    _Float16* keys2h = (_Float16*)alloc((size_t)Bc * NTXT * T2 * 2);   // (B,512,400)
    _Float16* R1h    = (_Float16*)alloc((size_t)Bc * 1024 * T2 * 2);   // kh (B,1024,400) / qh1 (B,160,2000)
    _Float16* kfh    = (_Float16*)alloc((size_t)Bc * NATT * T2 * 2);   // (B,80,400)
    _Float16* qh2h   = (_Float16*)alloc((size_t)Bc * NMEL * T1 * 2);   // (B,80,2000)
    _Float16* qfh    = (_Float16*)alloc((size_t)Bc * NATT * T1 * 2);   // (B,80,2000)
    _Float16* qfT    = (_Float16*)alloc((size_t)Bc * T1 * NATT * 2);   // (B,2000,80)
    _Float16* kfT    = (_Float16*)alloc((size_t)Bc * T2 * NATT * 2);   // (B,400,80)
    float*    q2     = (float*)alloc((size_t)Bc * T1 * 4);
    float*    k2     = (float*)alloc((size_t)Bc * T2 * 4);

    float* attn = (float*)d_out;
    float* lp   = attn + (size_t)Bc * T1 * T2;  // scores first, then logprob

    dim3 blk(256);

    // keys2 = f16(keys + style^T)
    addstyle_kernel<<<dim3((NTXT * T2 + 255) / 256, Bc), blk, 0, stream>>>(
        keys, style, keys2h, NTXT, T2);

    // G1: kh = relu(conv_k3(keys2))  M=1024 N=400 K=512 taps=3
    gemm_wmma<float, _Float16, _Float16, 3, 1, false, false>
        <<<dim3(7, 8, Bc), blk, 0, stream>>>(
        kw1, 0, NTXT * 3, 3,
        keys2h, NTXT * T2, T2,
        kb1, nullptr,
        R1h, 1024 * T2, T2,
        1024, T2, NTXT);

    // G2: kf = conv_k1(kh)  M=80 N=400 K=1024
    gemm_wmma<float, _Float16, _Float16, 1, 0, true, false>
        <<<dim3(7, 1, Bc), blk, 0, stream>>>(
        kw2, 0, 1024, 1,
        R1h, 1024 * T2, T2,
        kb2, nullptr,
        kfh, NATT * T2, T2,
        NATT, T2, 1024);

    // G3: qh1 = relu(conv_k3(queries))  M=160 N=2000 K=80 taps=3
    gemm_wmma<float, float, _Float16, 3, 1, false, false>
        <<<dim3(32, 2, Bc), blk, 0, stream>>>(
        qw1, 0, NMEL * 3, 3,
        queries, NMEL * T1, T1,
        qb1, nullptr,
        R1h, 160 * T1, T1,
        160, T1, NMEL);

    // G4: qh2 = relu(conv_k1(qh1))  M=80 N=2000 K=160
    gemm_wmma<float, _Float16, _Float16, 1, 1, true, false>
        <<<dim3(32, 1, Bc), blk, 0, stream>>>(
        qw2, 0, 160, 1,
        R1h, 160 * T1, T1,
        qb2, nullptr,
        qh2h, NMEL * T1, T1,
        NMEL, T1, 160);

    // G5: qf = conv_k1(qh2)  M=80 N=2000 K=80 (same instantiation as G2)
    gemm_wmma<float, _Float16, _Float16, 1, 0, true, false>
        <<<dim3(32, 1, Bc), blk, 0, stream>>>(
        qw3, 0, NMEL, 1,
        qh2h, NMEL * T1, T1,
        qb3, nullptr,
        qfh, NATT * T1, T1,
        NATT, T1, NMEL);

    // transposes for contiguous A-operand / sum-of-squares
    transpose_h_kernel<<<dim3(63, 3, Bc), dim3(32, 8), 0, stream>>>(
        qfh, qfT, NATT, T1, NATT * T1, T1 * NATT);
    transpose_h_kernel<<<dim3(13, 3, Bc), dim3(32, 8), 0, stream>>>(
        kfh, kfT, NATT, T2, NATT * T2, T2 * NATT);

    sumsq_t_kernel<<<dim3(8, Bc), blk, 0, stream>>>(qfT, q2, NATT, T1, T1 * NATT);
    sumsq_t_kernel<<<dim3(2, Bc), blk, 0, stream>>>(kfT, k2, NATT, T2, T2 * NATT);

    // scores = SCALE*(q2+k2-2*q^T k)  M=2000 N=400 K=80
    // A = qfT (f16, unit k-stride) staged via global_load_async_to_lds_b128
    gemm_wmma<_Float16, _Float16, float, 1, 2, true, true>
        <<<dim3(7, 16, Bc), blk, 0, stream>>>(
        qfT, T1 * NATT, NATT, 1,
        kfh, NATT * T2, T2,
        q2, k2,
        lp, T1 * T2, T2,
        T1, T2, NATT);

    // log-softmax + prior (in place) and masked softmax -> attn
    row_softmax_kernel<<<dim3(T1, Bc), dim3(128), 0, stream>>>(attn, lp, prior, mask, T1, T2);
}
